// Encoder_27616639713919
// MI455X (gfx1250) — matrix operand order repaired and checked by reading, not compiled, not run on hardware
//
#include <hip/hip_runtime.h>
#include <hip/hip_bf16.h>

typedef __bf16 bf16_t;
typedef __attribute__((ext_vector_type(16))) __bf16 v16bf;
typedef __attribute__((ext_vector_type(8)))  __bf16 v8bf;
typedef __attribute__((ext_vector_type(8)))  float  v8f;

#define GB 64      // batch
#define GT 512     // timesteps
#define GF 256     // features
#define GU 1024    // units
#define G3 (3*GU)  // 3072
#define GM (GB*GT) // 32768 rows of xp

#define NT3  (G3/16)   // 192 n-tiles over 3U
#define KTF  (GF/32)   // 8 k-tiles for input GEMM
#define KTU  (GU/32)   // 32 k-tiles for recurrent GEMM
#define MT   (GM/16)   // 2048 row-tiles of x
#define NG   4         // batch groups of 16
#define NW   8         // workgroups cooperating per group
#define HLEN (16*GU)   // packed h elements per group

// A-matrix packed index: (row 0..15, k) -> slot in packed tile stream.
__device__ __forceinline__ int a_packed_idx(int row, int k) {
    const int kt = k >> 5;
    const int kk = k & 31;
    const int lane = row + ((kk & 8) ? 16 : 0);
    const int e = (kk & 7) + ((kk & 16) ? 8 : 0);
    return (kt * 32 + lane) * 16 + e;
}

// ---------------------------------------------------------------------------
// Prep 1: pack kernel[F,3U] and W_rec[U,3U] fp32 -> bf16 WMMA-B tile layout.
// ---------------------------------------------------------------------------
__global__ void pack_w(const float* __restrict__ kern,
                       const float* __restrict__ wrec,
                       bf16_t* __restrict__ kb,
                       bf16_t* __restrict__ wb) {
    const int NK = NT3 * KTF * 32;
    const int NWC = NT3 * KTU * 32;
    int c = blockIdx.x * blockDim.x + threadIdx.x;
    if (c >= NK + NWC) return;
    const float* W; bf16_t* O; int KT;
    if (c < NK) { W = kern; O = kb; KT = KTF; }
    else        { c -= NK; W = wrec; O = wb; KT = KTU; }
    const int lane = c & 31;
    const int kt = (c >> 5) % KT;
    const int nt = (c >> 5) / KT;
    const int col = nt * 16 + (lane & 15);
    const int k0 = kt * 32 + (lane >> 4) * 8;
    __align__(32) bf16_t tmp[16];
#pragma unroll
    for (int e = 0; e < 16; ++e) tmp[e] = (bf16_t)W[(size_t)(k0 + e + (e >= 8 ? 8 : 0)) * G3 + col];
    *(v8bf*)(O + (size_t)c * 16)     = *(const v8bf*)&tmp[0];
    *(v8bf*)(O + (size_t)c * 16 + 8) = *(const v8bf*)&tmp[8];
}

// ---------------------------------------------------------------------------
// Prep 2: pack x[M,F] fp32 -> bf16 WMMA-A tile layout.
// ---------------------------------------------------------------------------
__global__ void pack_x(const float* __restrict__ x, bf16_t* __restrict__ xpk) {
    int c = blockIdx.x * blockDim.x + threadIdx.x;
    if (c >= MT * KTF * 32) return;
    const int lane = c & 31;
    const int kt = (c >> 5) % KTF;
    const int mt = (c >> 5) / KTF;
    const int row = mt * 16 + (lane & 15);
    const int kh = (lane >> 4) * 8;
    const float* xr = x + (size_t)row * GF + kt * 32;
    __align__(32) bf16_t tmp[16];
#pragma unroll
    for (int e = 0; e < 8; ++e) {
        tmp[e]     = (bf16_t)xr[kh + e];
        tmp[e + 8] = (bf16_t)xr[16 + kh + e];
    }
    *(v8bf*)(xpk + (size_t)c * 16)     = *(const v8bf*)&tmp[0];
    *(v8bf*)(xpk + (size_t)c * 16 + 8) = *(const v8bf*)&tmp[8];
}

// ---------------------------------------------------------------------------
// Prep 3: pack hidden state into hbuf[0] (A-layout per group); zero barriers.
// ---------------------------------------------------------------------------
__global__ void pack_h(const float* __restrict__ hidden, bf16_t* __restrict__ hbuf,
                       int* __restrict__ bar) {
    int i = blockIdx.x * blockDim.x + threadIdx.x;
    if (i < NG) bar[i] = 0;
    if (i >= NG * HLEN) return;
    const int g = i / HLEN;
    const int rem = i % HLEN;
    const int row = rem >> 10, k = rem & (GU - 1);
    hbuf[(size_t)g * HLEN + a_packed_idx(row, k)] =
        (bf16_t)hidden[(size_t)(g * 16 + row) * GU + k];
}

// ---------------------------------------------------------------------------
// Kernel 1: xp = x@kernel + b_in. Software-pipelined WMMA GEMM.
// ---------------------------------------------------------------------------
__global__ __launch_bounds__(256) void xp_gemm(const bf16_t* __restrict__ xpk,
                                               const bf16_t* __restrict__ kb,
                                               const float*  __restrict__ bias,
                                               float*        __restrict__ xp) {
    const int lane = threadIdx.x & 31;
    const int wave = threadIdx.x >> 5;
    const int mt   = blockIdx.y * 8 + wave;
    const int row0 = mt * 16;
    const int nt0  = blockIdx.x * 4;
    const int r    = lane & 15;
    const int hi   = (lane >> 4) * 8;
    const int lo   = lane * 16;

    const bf16_t* ap = xpk + (size_t)mt * KTF * 512;      // 512 = 32*16 per k-tile
    const bf16_t* bp[4];
#pragma unroll
    for (int nt = 0; nt < 4; ++nt) bp[nt] = kb + (size_t)(nt0 + nt) * KTF * 512;

    v8f acc[4] = {};
    v16bf a = *(const v16bf*)(ap + lo);
    v16bf b[4];
#pragma unroll
    for (int nt = 0; nt < 4; ++nt) b[nt] = *(const v16bf*)(bp[nt] + lo);

#pragma unroll
    for (int kt = 0; kt < KTF; ++kt) {
        const int kn = (kt + 1) & (KTF - 1);              // wraps; tail redundant
        const size_t off = (size_t)kn * 512 + lo;
        v16bf na = *(const v16bf*)(ap + off);
        v16bf nb[4];
#pragma unroll
        for (int nt = 0; nt < 4; ++nt) nb[nt] = *(const v16bf*)(bp[nt] + off);
#pragma unroll
        for (int nt = 0; nt < 4; ++nt)
            acc[nt] = __builtin_amdgcn_wmma_f32_16x16x32_bf16(
                false, a, false, b[nt], (short)0, acc[nt], false, false);
        a = na;
#pragma unroll
        for (int nt = 0; nt < 4; ++nt) b[nt] = nb[nt];
    }
#pragma unroll
    for (int nt = 0; nt < 4; ++nt) {
        const int col = (nt0 + nt) * 16 + r;
        const float bi = bias[col];
#pragma unroll
        for (int v = 0; v < 8; ++v) {
            const int row = row0 + hi + v;
            xp[(size_t)row * G3 + col] = acc[nt][v] + bi;
        }
    }
}

// ---------------------------------------------------------------------------
// Kernel 2: GRU recurrence. 4 groups x 8 cooperating WGs x 256 threads.
// Software-pipelined inner GEMM; ping-pong packed h; per-group device barrier.
// ---------------------------------------------------------------------------
__global__ __launch_bounds__(256) void gru_recurrent(
        bf16_t* __restrict__ hbuf,      // [2][NG][HLEN]
        int*    __restrict__ bar,       // [NG]
        const bf16_t* __restrict__ wb,
        const float*  __restrict__ bias,
        const float*  __restrict__ xp,
        float*        __restrict__ out) {
    __shared__ __align__(32) bf16_t hA[HLEN];   // 32 KB staged h

    const int g    = blockIdx.x / NW;
    const int w    = blockIdx.x % NW;
    const int lane = threadIdx.x & 31;
    const int wave = threadIdx.x >> 5;
    const int slice = w * 8 + wave;             // 0..63 -> u-slice of 16
    const int u    = slice * 16 + (lane & 15);
    const int hi   = (lane >> 4) * 8;
    const int lo   = lane * 16;

    const bf16_t* wz = wb + (size_t)(slice)       * KTU * 512;
    const bf16_t* wr = wb + (size_t)(64  + slice) * KTU * 512;
    const bf16_t* wh = wb + (size_t)(128 + slice) * KTU * 512;

    const float bz = bias[G3 + u];              // bias row 1 = b_rec
    const float br = bias[G3 + GU + u];
    const float bh = bias[G3 + 2 * GU + u];
    float* state = out + (size_t)GB * GT * GU;

    for (int t = 0; t < GT; ++t) {
        const bf16_t* hsrc = hbuf + ((size_t)(t & 1) * NG + g) * HLEN;
        bf16_t*       hdst = hbuf + ((size_t)((t + 1) & 1) * NG + g) * HLEN;

        // stage packed h into LDS (b128 copies)
        {
            const v8bf* s = (const v8bf*)hsrc;
            v8bf* d = (v8bf*)hA;
            for (int i = threadIdx.x; i < HLEN / 8; i += 256) d[i] = s[i];
        }
        __syncthreads();

        v8f az = {}, ar = {}, ah = {};
        v16bf a  = *(const v16bf*)(hA + lo);
        v16bf b0 = *(const v16bf*)(wz + lo);
        v16bf b1 = *(const v16bf*)(wr + lo);
        v16bf b2 = *(const v16bf*)(wh + lo);
        for (int kt = 0; kt < KTU; ++kt) {
            const int kn = (kt + 1) & (KTU - 1);          // wraps; tail redundant
            const size_t off = (size_t)kn * 512 + lo;
            v16bf na  = *(const v16bf*)(hA + off);
            v16bf nb0 = *(const v16bf*)(wz + off);
            v16bf nb1 = *(const v16bf*)(wr + off);
            v16bf nb2 = *(const v16bf*)(wh + off);
            az = __builtin_amdgcn_wmma_f32_16x16x32_bf16(false, a, false, b0, (short)0, az, false, false);
            ar = __builtin_amdgcn_wmma_f32_16x16x32_bf16(false, a, false, b1, (short)0, ar, false, false);
            ah = __builtin_amdgcn_wmma_f32_16x16x32_bf16(false, a, false, b2, (short)0, ah, false, false);
            a = na; b0 = nb0; b1 = nb1; b2 = nb2;
        }

        // fused gates for this wave's 16 u-columns
#pragma unroll
        for (int v = 0; v < 8; ++v) {
            const int row = hi + v;
            const size_t m = (size_t)(g * 16 + row) * GT + t;
            const size_t xb = m * G3;
            const float xz = xp[xb + u];
            const float xr = xp[xb + GU + u];
            const float xh = xp[xb + 2 * GU + u];
            const float rz = az[v] + bz;
            const float rr = ar[v] + br;
            const float rh = ah[v] + bh;
            const float z  = 1.f / (1.f + __expf(-(xz + rz)));
            const float rg = 1.f / (1.f + __expf(-(xr + rr)));
            const float hh = 1.f - 2.f / (__expf(2.f * (xh + rg * rh)) + 1.f);
            const float hold = (float)hA[a_packed_idx(row, u)];
            const float hn = z * hold + (1.f - z) * hh;
            out[m * GU + u] = hn;
            hdst[a_packed_idx(row, u)] = (bf16_t)hn;
            if (t == GT - 1) state[(size_t)(g * 16 + row) * GU + u] = hn;
        }

        // per-group device-scope barrier (release -> signal -> spin -> acquire)
        __threadfence();
        __syncthreads();
        if (threadIdx.x == 0) {
            atomicAdd(&bar[g], 1);
            const int target = NW * (t + 1);
            while (__hip_atomic_load(&bar[g], __ATOMIC_RELAXED,
                                     __HIP_MEMORY_SCOPE_AGENT) < target)
                __builtin_amdgcn_s_sleep(2);
        }
        __syncthreads();
        __threadfence();
    }
}

// ---------------------------------------------------------------------------
extern "C" void kernel_launch(void* const* d_in, const int* in_sizes, int n_in,
                              void* d_out, int out_size, void* d_ws, size_t ws_size,
                              hipStream_t stream) {
    const float* x      = (const float*)d_in[0];   // [B,T,F]
    const float* hidden = (const float*)d_in[1];   // [B,U]
    const float* kern   = (const float*)d_in[2];   // [F,3U]
    const float* wrec   = (const float*)d_in[3];   // [U,3U]
    const float* bias   = (const float*)d_in[4];   // [2,3U]
    float* out = (float*)d_out;

    // ws: xp fp32 | kern packed | wrec packed | x packed | hbuf x2 | barriers
    float*  xp     = (float*)d_ws;
    bf16_t* kern_b = (bf16_t*)(xp + (size_t)GM * G3);
    bf16_t* wrec_b = kern_b + (size_t)GF * G3;
    bf16_t* x_pk   = wrec_b + (size_t)GU * G3;
    bf16_t* hbuf   = x_pk + (size_t)MT * KTF * 32 * 16;
    int*    bar    = (int*)(hbuf + (size_t)2 * NG * HLEN);

    const int wchunks = NT3 * KTF * 32 + NT3 * KTU * 32;   // 245760
    pack_w<<<(wchunks + 255) / 256, 256, 0, stream>>>(kern, wrec, kern_b, wrec_b);

    const int xchunks = MT * KTF * 32;                      // 524288
    pack_x<<<(xchunks + 255) / 256, 256, 0, stream>>>(x, x_pk);

    pack_h<<<(NG * HLEN + 255) / 256, 256, 0, stream>>>(hidden, hbuf, bar);

    dim3 g1(NT3 / 4, MT / 8);                               // (48, 256)
    xp_gemm<<<g1, 256, 0, stream>>>(x_pk, kern_b, bias, xp);

    gru_recurrent<<<NG * NW, 256, 0, stream>>>(hbuf, bar, wrec_b, bias, xp, out);
}
